// My_Seg_72610717106312
// MI455X (gfx1250) — matrix operand order repaired and checked by reading, not compiled, not run on hardware
//
#include <hip/hip_runtime.h>
#include <hip/hip_bf16.h>
#include <initializer_list>

#define KNN_K 20
#define BB    2
#define NPTS  6000
#define S1    (BB*NPTS)          // 12000 point-samples
#define SEDG  (S1*KNN_K)         // 240000 edge-samples

typedef __attribute__((ext_vector_type(16))) _Float16 v16h;
typedef __attribute__((ext_vector_type(8)))  _Float16 v8h;
typedef __attribute__((ext_vector_type(8)))  float    v8f;

static inline unsigned cdiv_u(size_t a, size_t b) { return (unsigned)((a + b - 1) / b); }

// ---------------------------------------------------------------------------
// f32 -> f16 weight conversion
// ---------------------------------------------------------------------------
__global__ void k_cvt16(const float* __restrict__ a, _Float16* __restrict__ o, int n) {
  int i = blockIdx.x * 256 + threadIdx.x;
  if (i < n) o[i] = (_Float16)a[i];
}

// ---------------------------------------------------------------------------
// WMMA GEMM: Y[s, o] = sum_c W[o, c] * X[s, c]
// W: (Cout x Cin) f16 row-major.  X: (S x Cin) f16 row-major.  Y: (S x Cout) f16.
// One 16(sample) x 16(out) tile per wave, K-loop in steps of 32.
// A layout (16x32 f16): lane m=L&15 -> row M=m; VGPR r pair at K=16*half+2r
// B layout (32x16 f16): lane -> col N=L&15; VGPR r pair at K=16*half+2r
// D layout (16x16 f32): lane -> col N=L&15; VGPR r -> row M=r+8*half
// ---------------------------------------------------------------------------
__global__ void k_wmma_gemm(const _Float16* __restrict__ W, const _Float16* __restrict__ X,
                            _Float16* __restrict__ Y, int S, int Cin, int Cout)
{
  const int lane = threadIdx.x & 31;
  const int wv   = threadIdx.x >> 5;
  const int m    = lane & 15, hf = lane >> 4;
  const int s0   = (blockIdx.x * 8 + wv) * 16;
  if (s0 >= S) return;                       // wave-uniform exit, EXEC stays all-ones
  const int oBase = blockIdx.y * 16;
  v8f acc = {};
  for (int k0 = 0; k0 < Cin; k0 += 32) {
    union { v16h v; unsigned u[8]; } a, b;
    const unsigned* wrow = (const unsigned*)(W + (size_t)(oBase + m) * Cin + k0);
#pragma unroll
    for (int r = 0; r < 8; ++r)
      a.u[r] = wrow[r + 8 * hf];
    const unsigned* xrow = (const unsigned*)(X + (size_t)(s0 + m) * Cin + k0 + 16 * hf);
#pragma unroll
    for (int r = 0; r < 8; ++r) b.u[r] = xrow[r];
    acc = __builtin_amdgcn_wmma_f32_16x16x32_f16(false, a.v, false, b.v, (short)0, acc,
                                                 false, false);
  }
  v8h o;
#pragma unroll
  for (int r = 0; r < 8; ++r) o[r] = (_Float16)acc[r];
  *(v8h*)(Y + (size_t)(s0 + m) * Cout + oBase + 8 * hf) = o;
}

// ---------------------------------------------------------------------------
// WMMA GEMM with fused edge-feature gather.
// Logical B matrix row s (edge sample) has Cin=2C channels:
//   c <  C : F[b*N + idx[s]][c]    (neighbor)
//   c >= C : F[b*N + n     ][c-C]  (center)
// C is a multiple of 32, so each 32-wide K-tile is uniformly neighbor or center.
// ---------------------------------------------------------------------------
__global__ void k_wmma_gemm_edge(const _Float16* __restrict__ W, const _Float16* __restrict__ F,
                                 const int* __restrict__ idx, _Float16* __restrict__ Y,
                                 int N, int C, int Cout)
{
  const int lane = threadIdx.x & 31;
  const int wv   = threadIdx.x >> 5;
  const int m    = lane & 15, hf = lane >> 4;
  const int s0   = (blockIdx.x * 8 + wv) * 16;
  const int SE   = BB * N * KNN_K;
  if (s0 >= SE) return;
  const int s  = s0 + m;
  const int b  = s / (N * KNN_K);
  const int r0 = s % (N * KNN_K);
  const int n  = r0 / KNN_K;
  const size_t rowN = (size_t)b * N + idx[s];
  const size_t rowC = (size_t)b * N + n;
  const int Cin   = 2 * C;
  const int oBase = blockIdx.y * 16;
  v8f acc = {};
  for (int k0 = 0; k0 < Cin; k0 += 32) {
    union { v16h v; unsigned u[8]; } a, bb;
    const unsigned* wrow = (const unsigned*)(W + (size_t)(oBase + m) * Cin + k0);
#pragma unroll
    for (int r = 0; r < 8; ++r)
      a.u[r] = wrow[r + 8 * hf];
    const bool useN = (k0 < C);
    const size_t row = useN ? rowN : rowC;
    const int col = k0 - (useN ? 0 : C) + 16 * hf;
    const unsigned* xrow = (const unsigned*)(F + row * C + col);
#pragma unroll
    for (int r = 0; r < 8; ++r) bb.u[r] = xrow[r];
    acc = __builtin_amdgcn_wmma_f32_16x16x32_f16(false, a.v, false, bb.v, (short)0, acc,
                                                 false, false);
  }
  v8h o;
#pragma unroll
  for (int r = 0; r < 8; ++r) o[r] = (_Float16)acc[r];
  *(v8h*)(Y + (size_t)s * Cout + oBase + 8 * hf) = o;
}

// ---------------------------------------------------------------------------
// BatchNorm stats over all rows per channel: scale = g*rsqrt(var+eps),
// shift = b - mean*scale.  One block per channel (preact is L2-resident).
// ---------------------------------------------------------------------------
__global__ void k_bn_stats(const _Float16* __restrict__ Y, const float* __restrict__ g,
                           const float* __restrict__ bt, float* __restrict__ scale,
                           float* __restrict__ shift, int S, int C)
{
  __shared__ float r1[256], r2[256];
  const int c = blockIdx.x;
  float s = 0.f, s2 = 0.f;
  for (int i = threadIdx.x; i < S; i += 256) {
    float v = (float)Y[(size_t)i * C + c];
    s += v; s2 += v * v;
  }
  r1[threadIdx.x] = s; r2[threadIdx.x] = s2;
  __syncthreads();
  for (int o = 128; o > 0; o >>= 1) {
    if (threadIdx.x < o) { r1[threadIdx.x] += r1[threadIdx.x + o]; r2[threadIdx.x] += r2[threadIdx.x + o]; }
    __syncthreads();
  }
  if (threadIdx.x == 0) {
    float mean = r1[0] / S;
    float var  = r2[0] / S - mean * mean;
    float inv  = rsqrtf(var + 1e-5f);
    float sc   = g[c] * inv;
    scale[c] = sc;
    shift[c] = bt[c] - mean * sc;
  }
}

// act: 0 = relu, 1 = leaky-relu(0.2)
__global__ void k_bn_act(const _Float16* __restrict__ Y, const float* __restrict__ sc,
                         const float* __restrict__ sh, _Float16* __restrict__ out,
                         size_t tot, int C, int act)
{
  size_t i = (size_t)blockIdx.x * 256 + threadIdx.x;
  if (i >= tot) return;
  int c = (int)(i % C);
  float v = (float)Y[i] * sc[c] + sh[c];
  v = (act == 0) ? fmaxf(v, 0.f) : (v > 0.f ? v : 0.2f * v);
  out[i] = (_Float16)v;
}

// BN + leaky-relu + max over K edge slots: (S1*K x C) -> (S1 x C)
__global__ void k_bn_lrelu_maxk(const _Float16* __restrict__ Y, const float* __restrict__ sc,
                                const float* __restrict__ sh, _Float16* __restrict__ out,
                                int Srows, int C)
{
  size_t i = (size_t)blockIdx.x * 256 + threadIdx.x;
  if (i >= (size_t)Srows * C) return;
  int s1 = (int)(i / C), c = (int)(i % C);
  float a = sc[c], b = sh[c];
  float mx = -3.4e38f;
  for (int k = 0; k < KNN_K; ++k) {
    float v = (float)Y[((size_t)s1 * KNN_K + k) * C + c] * a + b;
    v = v > 0.f ? v : 0.2f * v;
    mx = fmaxf(mx, v);
  }
  out[i] = (_Float16)mx;
}

// ---------------------------------------------------------------------------
// Fused kNN: one wave per query. Per-lane top-20 lists in LDS, final 20-round
// cross-lane merge via shuffles. Emits the k-nearest SET (order-free; only
// max-over-K consumes it). f layout: (B*N x ldf) f16.
// ---------------------------------------------------------------------------
__global__ void k_knn(const _Float16* __restrict__ f, int* __restrict__ idxout,
                      int N, int C, int ldf)
{
  __shared__ float    sd[8][32][KNN_K];
  __shared__ int      si[8][32][KNN_K];
  __shared__ _Float16 qf[8][128];
  const int lane = threadIdx.x & 31;
  const int wv   = threadIdx.x >> 5;
  const int gw   = blockIdx.x * 8 + wv;     // query id in [0, B*N)
  const int b    = gw / N, q = gw % N;
  const _Float16* base = f + (size_t)b * N * ldf;
  for (int c = lane; c < C; c += 32) qf[wv][c] = base[(size_t)q * ldf + c];
  for (int j = 0; j < KNN_K; ++j) { sd[wv][lane][j] = 1e30f; si[wv][lane][j] = 0; }
  __syncthreads();
  float worst = 1e30f; int worstPos = 0;
  for (int m0 = 0; m0 < N; m0 += 32) {
    const int m = m0 + lane;
    float d = 1e30f;
    if (m < N) {
      const _Float16* fr = base + (size_t)m * ldf;
      d = 0.f;
      for (int c = 0; c < C; ++c) { float t = (float)fr[c] - (float)qf[wv][c]; d += t * t; }
    }
    if (d < worst) {
      sd[wv][lane][worstPos] = d; si[wv][lane][worstPos] = m;
      worst = -1e30f;
      for (int j = 0; j < KNN_K; ++j) {
        float v = sd[wv][lane][j];
        if (v > worst) { worst = v; worstPos = j; }
      }
    }
  }
  __syncthreads();
  const int outbase = gw * KNN_K;
  for (int r = 0; r < KNN_K; ++r) {
    float bm = 1e30f; int bp = 0;
    for (int j = 0; j < KNN_K; ++j) {
      float v = sd[wv][lane][j];
      if (v < bm) { bm = v; bp = j; }
    }
    float rv = bm; int rl = lane;
    for (int off = 16; off > 0; off >>= 1) {
      float ov = __shfl_xor(rv, off, 32);
      int   ol = __shfl_xor(rl, off, 32);
      if (ov < rv || (ov == rv && ol < rl)) { rv = ov; rl = ol; }
    }
    int wp   = __shfl(bp, rl, 32);
    int widx = si[wv][rl][wp];
    if (lane == 0) idxout[outbase + r] = widx;
    if (lane == rl) sd[wv][lane][wp] = 1e30f;
  }
}

// ---------------------------------------------------------------------------
// Small VALU kernels
// ---------------------------------------------------------------------------
// STN conv1: 3 -> 64, reads x (B,6,N) f32 directly, writes preact f16 (S1 x 64)
__global__ void k_stn_conv1(const float* __restrict__ x, const float* __restrict__ W,
                            _Float16* __restrict__ Y, int N, int coff)
{
  int i = blockIdx.x * 256 + threadIdx.x;
  if (i >= BB * N) return;
  int b = i / N, n = i % N;
  const float* xb = x + (size_t)b * 6 * N + (size_t)coff * N + n;
  float v0 = xb[0], v1 = xb[N], v2 = xb[2 * N];
  _Float16* y = Y + (size_t)i * 64;
  for (int o = 0; o < 64; ++o)
    y[o] = (_Float16)(W[o * 3] * v0 + W[o * 3 + 1] * v1 + W[o * 3 + 2] * v2);
}

// max over N per (batch, channel): (B*N x C) f16 -> (B x C) f32
__global__ void k_maxpool(const _Float16* __restrict__ a, float* __restrict__ out, int Nn, int C)
{
  int i = blockIdx.x * 256 + threadIdx.x;
  if (i >= BB * C) return;
  int b = i / C, c = i % C;
  float m = -3.4e38f;
  for (int n = 0; n < Nn; ++n)
    m = fmaxf(m, (float)a[((size_t)b * Nn + n) * C + c]);
  out[i] = m;
}

// dense FC on tiny batch: out[b,o] = act(W[o,:]·in[b,:] + bias[o])
__global__ void k_fc(const float* __restrict__ in, const float* __restrict__ W,
                     const float* __restrict__ bias, float* __restrict__ out,
                     int Bn, int Cin, int Cout, int relu)
{
  int i = blockIdx.x * 256 + threadIdx.x;
  if (i >= Bn * Cout) return;
  int b = i / Cout, o = i % Cout;
  float a = bias[o];
  const float* wr = W + (size_t)o * Cin;
  const float* xr = in + (size_t)b * Cin;
  for (int c = 0; c < Cin; ++c) a += wr[c] * xr[c];
  if (relu) a = fmaxf(a, 0.f);
  out[i] = a;
}

// apply (fc3 + I3) transform: out[(b*N+n)*4 + d] = sum_c x[b, coff+c, n] * T[b][c][d]
__global__ void k_transform3(const float* __restrict__ x, const float* __restrict__ t9,
                             _Float16* __restrict__ out, int N, int coff)
{
  int i = blockIdx.x * 256 + threadIdx.x;
  if (i >= BB * N) return;
  int b = i / N, n = i % N;
  const float* xb = x + (size_t)b * 6 * N + (size_t)coff * N + n;
  float v0 = xb[0], v1 = xb[N], v2 = xb[2 * N];
  const float* T = t9 + b * 9;
  for (int d = 0; d < 3; ++d) {
    float r = v0 * (T[d] + (d == 0 ? 1.f : 0.f))
            + v1 * (T[3 + d] + (d == 1 ? 1.f : 0.f))
            + v2 * (T[6 + d] + (d == 2 ? 1.f : 0.f));
    out[(size_t)i * 4 + d] = (_Float16)r;
  }
  out[(size_t)i * 4 + 3] = (_Float16)0.f;
}

// edge conv1: Cin = 2*3 (neighbor || center), Cout = 64, W f32 (64x6)
__global__ void k_edge_conv6(const _Float16* __restrict__ f, const int* __restrict__ idx,
                             const float* __restrict__ W, _Float16* __restrict__ Y, int N)
{
  __shared__ float Ws[64 * 6];
  for (int i = threadIdx.x; i < 64 * 6; i += 256) Ws[i] = W[i];
  __syncthreads();
  int s = blockIdx.x * 256 + threadIdx.x;
  if (s >= BB * N * KNN_K) return;
  int b = s / (N * KNN_K), r = s % (N * KNN_K), n = r / KNN_K;
  int nb = idx[s];
  const _Float16* fn = f + ((size_t)b * N + nb) * 4;
  const _Float16* fc = f + ((size_t)b * N + n) * 4;
  float in[6] = { (float)fn[0], (float)fn[1], (float)fn[2],
                  (float)fc[0], (float)fc[1], (float)fc[2] };
  _Float16* y = Y + (size_t)s * 64;
  for (int o = 0; o < 64; ++o) {
    float a = 0.f;
#pragma unroll
    for (int c = 0; c < 6; ++c) a += Ws[o * 6 + c] * in[c];
    y[o] = (_Float16)a;
  }
}

// channel-block copy for concatenation
__global__ void k_copy_cols(const _Float16* __restrict__ src, _Float16* __restrict__ dst,
                            int S, int Csrc, int Cdst, int off)
{
  size_t i = (size_t)blockIdx.x * 256 + threadIdx.x;
  if (i >= (size_t)S * Csrc) return;
  int s = (int)(i / Csrc), c = (int)(i % Csrc);
  dst[(size_t)s * Cdst + off + c] = src[i];
}

// final head: out[b,o,n] = p3w[o,:]·h2[b*N+n,:] + p3b[o]   (f32, no BN/act)
__global__ void k_final(const _Float16* __restrict__ h2, const float* __restrict__ W,
                        const float* __restrict__ bias, float* __restrict__ out, int N)
{
  int i = blockIdx.x * 256 + threadIdx.x;
  if (i >= BB * 15 * N) return;
  int b = i / (15 * N), r = i % (15 * N), o = r / N, n = r % N;
  const _Float16* h = h2 + ((size_t)b * N + n) * 256;
  const float* wr = W + (size_t)o * 256;
  float a = bias[o];
  for (int c = 0; c < 256; ++c) a += wr[c] * (float)h[c];
  out[i] = a;
}

// ---------------------------------------------------------------------------
// Host orchestration
// ---------------------------------------------------------------------------
struct BranchIdx {
  int sw1, sg1, sb1, sw2, sg2, sb2, sw3, sg3, sb3, fc1w, fc1b, fc2w, fc2b, fc3w, fc3b;
  int w1, g1, b1, w2, g2, b2, w3, g3, b3, w5, g5, b5;
};
struct NetIdx { int x; BranchIdx c, n; int p1w, p1g, p1b, p2w, p2g, p2b, p3w, p3b; };

static void fill_ins(BranchIdx& b, int& k) {
  b.sw1 = k++; b.sg1 = k++; b.sb1 = k++; b.sw2 = k++; b.sg2 = k++; b.sb2 = k++;
  b.sw3 = k++; b.sg3 = k++; b.sb3 = k++;
  b.fc1w = k++; b.fc1b = k++; b.fc2w = k++; b.fc2b = k++; b.fc3w = k++; b.fc3b = k++;
  b.w1 = k++; b.g1 = k++; b.b1 = k++; b.w2 = k++; b.g2 = k++; b.b2 = k++;
  b.w3 = k++; b.g3 = k++; b.b3 = k++; b.w5 = k++; b.g5 = k++; b.b5 = k++;
}
static void fill_sorted(BranchIdx& b, int base) {
  b.b1 = base + 0;  b.b2 = base + 1;  b.b3 = base + 2;  b.b5 = base + 3;
  b.g1 = base + 4;  b.g2 = base + 5;  b.g3 = base + 6;  b.g5 = base + 7;
  b.sb1 = base + 8; b.sb2 = base + 9; b.sb3 = base + 10;
  b.fc1b = base + 11; b.fc1w = base + 12; b.fc2b = base + 13; b.fc2w = base + 14;
  b.fc3b = base + 15; b.fc3w = base + 16;
  b.sg1 = base + 17; b.sg2 = base + 18; b.sg3 = base + 19;
  b.sw1 = base + 20; b.sw2 = base + 21; b.sw3 = base + 22;
  b.w1 = base + 23; b.w2 = base + 24; b.w3 = base + 25; b.w5 = base + 26;
}

extern "C" void kernel_launch(void* const* d_in, const int* in_sizes, int n_in,
                              void* d_out, int out_size, void* d_ws, size_t ws_size,
                              hipStream_t stream)
{
  (void)n_in; (void)out_size; (void)ws_size;
  // insertion-order DFS (x first) vs jax tree_flatten sorted-keys (x last)
  const bool xfirst = (in_sizes[0] == 72000);
  NetIdx I;
  if (xfirst) {
    I.x = 0; int k = 1;
    fill_ins(I.c, k); fill_ins(I.n, k);
    I.p1w = k++; I.p1g = k++; I.p1b = k++; I.p2w = k++; I.p2g = k++; I.p2b = k++;
    I.p3w = k++; I.p3b = k++;
  } else {
    fill_sorted(I.c, 0); fill_sorted(I.n, 27);
    I.p1b = 54; I.p1g = 55; I.p1w = 56; I.p2b = 57; I.p2g = 58; I.p2w = 59;
    I.p3b = 60; I.p3w = 61; I.x = 62;
  }
  auto F = [&](int i) { return (const float*)d_in[i]; };

  char* wsb = (char*)d_ws;
  size_t off = 0;
  auto alloc = [&](size_t bytes) { size_t o = off; off += (bytes + 255) & ~(size_t)255; return o; };
  // f16 weight copies
  size_t o_cstn_w2 = alloc(8192 * 2),   o_cstn_w3 = alloc(131072 * 2);
  size_t o_nstn_w2 = alloc(8192 * 2),   o_nstn_w3 = alloc(131072 * 2);
  size_t o_c_w2 = alloc(16384 * 2), o_c_w3 = alloc(65536 * 2), o_c_w5 = alloc(114688 * 2);
  size_t o_n_w2 = alloc(16384 * 2), o_n_w3 = alloc(65536 * 2), o_n_w5 = alloc(114688 * 2);
  size_t o_p1w = alloc(262144 * 2), o_p2w = alloc(131072 * 2);
  // activations / scratch
  size_t o_coorT = alloc((size_t)S1 * 4 * 2), o_norT = alloc((size_t)S1 * 4 * 2);
  size_t o_stnA = alloc((size_t)S1 * 64 * 2), o_stnB = alloc((size_t)S1 * 128 * 2);
  size_t o_stnC = alloc((size_t)S1 * 1024 * 2);
  size_t o_pool = alloc(2 * 1024 * 4), o_fcb1 = alloc(2 * 512 * 4);
  size_t o_fcb2 = alloc(2 * 256 * 4), o_t9 = alloc(2 * 9 * 4);
  size_t o_scale = alloc(1024 * 4), o_shift = alloc(1024 * 4);
  size_t o_idx = alloc((size_t)SEDG * 4);
  size_t o_c1 = alloc((size_t)S1 * 64 * 2),  o_n1 = alloc((size_t)S1 * 64 * 2);
  size_t o_c2 = alloc((size_t)S1 * 128 * 2), o_n2 = alloc((size_t)S1 * 128 * 2);
  size_t o_c3 = alloc((size_t)S1 * 256 * 2), o_n3 = alloc((size_t)S1 * 256 * 2);
  size_t o_cf = alloc((size_t)S1 * 256 * 2), o_nf = alloc((size_t)S1 * 256 * 2);
  size_t o_cat448 = alloc((size_t)S1 * 448 * 2), o_cat512 = alloc((size_t)S1 * 512 * 2);
  size_t o_h1 = alloc((size_t)S1 * 512 * 2), o_h2 = alloc((size_t)S1 * 256 * 2);
  size_t o_pre = alloc((size_t)SEDG * 256 * 2);   // 123 MB shared preact buffer

  auto H  = [&](size_t o) { return (_Float16*)(wsb + o); };
  auto FP = [&](size_t o) { return (float*)(wsb + o); };
  auto IP = [&](size_t o) { return (int*)(wsb + o); };

  auto cvt = [&](int src, size_t dst, int n) {
    k_cvt16<<<cdiv_u(n, 256), 256, 0, stream>>>(F(src), H(dst), n);
  };
  cvt(I.c.sw2, o_cstn_w2, 8192);  cvt(I.c.sw3, o_cstn_w3, 131072);
  cvt(I.n.sw2, o_nstn_w2, 8192);  cvt(I.n.sw3, o_nstn_w3, 131072);
  cvt(I.c.w2, o_c_w2, 16384); cvt(I.c.w3, o_c_w3, 65536); cvt(I.c.w5, o_c_w5, 114688);
  cvt(I.n.w2, o_n_w2, 16384); cvt(I.n.w3, o_n_w3, 65536); cvt(I.n.w5, o_n_w5, 114688);
  cvt(I.p1w, o_p1w, 262144); cvt(I.p2w, o_p2w, 131072);

  // plain conv layer: WMMA GEMM -> BN stats -> BN + act (0 relu, 1 lrelu)
  auto plain = [&](size_t wo, const _Float16* Xin, int S, int Cin, int Cout,
                   int gi, int bi, _Float16* Aout, int act) {
    dim3 g(cdiv_u(S / 16, 8), Cout / 16);
    k_wmma_gemm<<<g, 256, 0, stream>>>(H(wo), Xin, H(o_pre), S, Cin, Cout);
    k_bn_stats<<<Cout, 256, 0, stream>>>(H(o_pre), F(gi), F(bi), FP(o_scale), FP(o_shift), S, Cout);
    size_t tot = (size_t)S * Cout;
    k_bn_act<<<cdiv_u(tot, 256), 256, 0, stream>>>(H(o_pre), FP(o_scale), FP(o_shift),
                                                   Aout, tot, Cout, act);
  };

  // STN for one branch -> t9 (2x9 f32, without identity; added in transform)
  auto stn = [&](const BranchIdx& br, size_t w2o, size_t w3o, int coff) {
    k_stn_conv1<<<cdiv_u(S1, 256), 256, 0, stream>>>(F(I.x), F(br.sw1), H(o_pre), NPTS, coff);
    k_bn_stats<<<64, 256, 0, stream>>>(H(o_pre), F(br.sg1), F(br.sb1),
                                       FP(o_scale), FP(o_shift), S1, 64);
    k_bn_act<<<cdiv_u((size_t)S1 * 64, 256), 256, 0, stream>>>(
        H(o_pre), FP(o_scale), FP(o_shift), H(o_stnA), (size_t)S1 * 64, 64, 0);
    plain(w2o, H(o_stnA), S1, 64, 128, br.sg2, br.sb2, H(o_stnB), 0);
    plain(w3o, H(o_stnB), S1, 128, 1024, br.sg3, br.sb3, H(o_stnC), 0);
    k_maxpool<<<cdiv_u(2 * 1024, 256), 256, 0, stream>>>(H(o_stnC), FP(o_pool), NPTS, 1024);
    k_fc<<<cdiv_u(2 * 512, 256), 256, 0, stream>>>(FP(o_pool), F(br.fc1w), F(br.fc1b),
                                                   FP(o_fcb1), 2, 1024, 512, 1);
    k_fc<<<cdiv_u(2 * 256, 256), 256, 0, stream>>>(FP(o_fcb1), F(br.fc2w), F(br.fc2b),
                                                   FP(o_fcb2), 2, 512, 256, 1);
    k_fc<<<1, 256, 0, stream>>>(FP(o_fcb2), F(br.fc3w), F(br.fc3b), FP(o_t9), 2, 256, 9, 0);
  };

  stn(I.c, o_cstn_w2, o_cstn_w3, 0);
  k_transform3<<<cdiv_u(S1, 256), 256, 0, stream>>>(F(I.x), FP(o_t9), H(o_coorT), NPTS, 0);
  stn(I.n, o_nstn_w2, o_nstn_w3, 3);
  k_transform3<<<cdiv_u(S1, 256), 256, 0, stream>>>(F(I.x), FP(o_t9), H(o_norT), NPTS, 3);

  // edge conv stage (WMMA path): gather-fused GEMM -> stats -> BN+lrelu+maxK
  auto edge = [&](size_t wo, const _Float16* fin, int C, int Cout,
                  int gi, int bi, _Float16* Aout) {
    dim3 g(cdiv_u(SEDG / 16, 8), Cout / 16);
    k_wmma_gemm_edge<<<g, 256, 0, stream>>>(H(wo), fin, IP(o_idx), H(o_pre), NPTS, C, Cout);
    k_bn_stats<<<Cout, 256, 0, stream>>>(H(o_pre), F(gi), F(bi),
                                         FP(o_scale), FP(o_shift), SEDG, Cout);
    k_bn_lrelu_maxk<<<cdiv_u((size_t)S1 * Cout, 256), 256, 0, stream>>>(
        H(o_pre), FP(o_scale), FP(o_shift), Aout, S1, Cout);
  };

  // stage 1: kNN on transformed coords (C=3, ld=4), Cin=6 edge conv (VALU)
  k_knn<<<S1 / 8, 256, 0, stream>>>(H(o_coorT), IP(o_idx), NPTS, 3, 4);
  k_edge_conv6<<<cdiv_u(SEDG, 256), 256, 0, stream>>>(H(o_coorT), IP(o_idx), F(I.c.w1),
                                                      H(o_pre), NPTS);
  k_bn_stats<<<64, 256, 0, stream>>>(H(o_pre), F(I.c.g1), F(I.c.b1),
                                     FP(o_scale), FP(o_shift), SEDG, 64);
  k_bn_lrelu_maxk<<<cdiv_u((size_t)S1 * 64, 256), 256, 0, stream>>>(
      H(o_pre), FP(o_scale), FP(o_shift), H(o_c1), S1, 64);
  k_edge_conv6<<<cdiv_u(SEDG, 256), 256, 0, stream>>>(H(o_norT), IP(o_idx), F(I.n.w1),
                                                      H(o_pre), NPTS);
  k_bn_stats<<<64, 256, 0, stream>>>(H(o_pre), F(I.n.g1), F(I.n.b1),
                                     FP(o_scale), FP(o_shift), SEDG, 64);
  k_bn_lrelu_maxk<<<cdiv_u((size_t)S1 * 64, 256), 256, 0, stream>>>(
      H(o_pre), FP(o_scale), FP(o_shift), H(o_n1), S1, 64);

  // stage 2
  k_knn<<<S1 / 8, 256, 0, stream>>>(H(o_c1), IP(o_idx), NPTS, 64, 64);
  edge(o_c_w2, H(o_c1), 64, 128, I.c.g2, I.c.b2, H(o_c2));
  edge(o_n_w2, H(o_n1), 64, 128, I.n.g2, I.n.b2, H(o_n2));

  // stage 3
  k_knn<<<S1 / 8, 256, 0, stream>>>(H(o_c2), IP(o_idx), NPTS, 128, 128);
  edge(o_c_w3, H(o_c2), 128, 256, I.c.g3, I.c.b3, H(o_c3));
  edge(o_n_w3, H(o_n2), 128, 256, I.n.g3, I.n.b3, H(o_n3));

  // concat(c1,c2,c3) -> w5 -> cf ; same for normals branch
  auto cat3 = [&](size_t a, size_t b2o, size_t c3o) {
    k_copy_cols<<<cdiv_u((size_t)S1 * 64, 256), 256, 0, stream>>>(H(a), H(o_cat448), S1, 64, 448, 0);
    k_copy_cols<<<cdiv_u((size_t)S1 * 128, 256), 256, 0, stream>>>(H(b2o), H(o_cat448), S1, 128, 448, 64);
    k_copy_cols<<<cdiv_u((size_t)S1 * 256, 256), 256, 0, stream>>>(H(c3o), H(o_cat448), S1, 256, 448, 192);
  };
  cat3(o_c1, o_c2, o_c3);
  plain(o_c_w5, H(o_cat448), S1, 448, 256, I.c.g5, I.c.b5, H(o_cf), 1);
  cat3(o_n1, o_n2, o_n3);
  plain(o_n_w5, H(o_cat448), S1, 448, 256, I.n.g5, I.n.b5, H(o_nf), 1);

  // concat(cf, nf) -> p1 -> p2 -> head
  k_copy_cols<<<cdiv_u((size_t)S1 * 256, 256), 256, 0, stream>>>(H(o_cf), H(o_cat512), S1, 256, 512, 0);
  k_copy_cols<<<cdiv_u((size_t)S1 * 256, 256), 256, 0, stream>>>(H(o_nf), H(o_cat512), S1, 256, 512, 256);
  plain(o_p1w, H(o_cat512), S1, 512, 512, I.p1g, I.p1b, H(o_h1), 1);
  plain(o_p2w, H(o_h1),     S1, 512, 256, I.p2g, I.p2b, H(o_h2), 1);

  k_final<<<cdiv_u(BB * 15 * NPTS, 256), 256, 0, stream>>>(H(o_h2), F(I.p3w), F(I.p3b),
                                                           (float*)d_out, NPTS);
}
